// RNN_73658689126422
// MI455X (gfx1250) — matrix operand order repaired and checked by reading, not compiled, not run on hardware
//
#include <hip/hip_runtime.h>

typedef __attribute__((ext_vector_type(16))) _Float16 v16h;
typedef __attribute__((ext_vector_type(8)))  _Float16 v8h;
typedef __attribute__((ext_vector_type(8)))  float    v8f;

constexpr int SEQ   = 1024;
constexpr int BATCH = 256;
constexpr int K_IN  = 6;
constexpr int DIM   = 512;
constexpr int TPB   = 256;   // 8 waves (wave32)
constexpr int NWG   = 32;    // 4 M-blocks x 8 N-blocks

__device__ __forceinline__ float fast_tanh(float v) {
    // tanh(v) = 1 - 2/(exp(2v)+1); exact at +/-inf saturation
    float e = __expf(2.0f * v);
    return 1.0f - 2.0f / (e + 1.0f);
}

__global__ void rnn_init(float* __restrict__ y, const float* __restrict__ fc_b,
                         const float* __restrict__ hidden,
                         _Float16* __restrict__ h0, unsigned* __restrict__ cnt) {
    int i = blockIdx.x * blockDim.x + threadIdx.x;
    if (i < SEQ * BATCH) y[i] = fc_b[0];
    if (i < BATCH * DIM) h0[i] = (_Float16)hidden[i];
    if (i < SEQ)         cnt[i] = 0u;
}

__launch_bounds__(TPB, 1)
__global__ void rnn_persistent(const float* __restrict__ x,
                               const float* __restrict__ Wih,
                               const float* __restrict__ Whh,
                               const float* __restrict__ b_ih,
                               const float* __restrict__ b_hh,
                               const float* __restrict__ fc_w,
                               float* __restrict__ out,      // [SEQ*BATCH] y, then [BATCH*DIM] h_last
                               _Float16* __restrict__ h0,
                               _Float16* __restrict__ h1,
                               unsigned* __restrict__ cnt) {
    // W_hh slice for this WG's 64 output columns, f16, XOR-swizzled rows: 64 KB
    __shared__ __align__(16) _Float16 sW[64 * 512];

    const int tid  = threadIdx.x;
    const int wg   = blockIdx.x;
    const int mblk = wg & 3;      // 4 blocks of 64 batch rows
    const int nblk = wg >> 2;     // 8 blocks of 64 hidden cols

    // Stage W_hh[n, k] (row-major, n = output col) into LDS as f16 with XOR swizzle.
    for (int idx = tid; idx < 64 * 512; idx += TPB) {
        int nl = idx >> 9;
        int k  = idx & 511;
        _Float16 v = (_Float16)Whh[(size_t)(nblk * 64 + nl) * DIM + k];
        sW[(nl << 9) | (k ^ ((nl & 15) << 3))] = v;
    }
    __syncthreads();

    const int wave = tid >> 5;
    const int lane = tid & 31;
    const int wm   = wave & 3;           // 4 M sub-blocks of 16 rows
    const int wn   = wave >> 2;          // 2 N sub-blocks of 32 cols
    const int m0   = mblk * 64 + wm * 16;
    const int hsel = lane >> 4;          // lane half (0/1)
    const int l16  = lane & 15;

    const int nl0 = wn * 32 + l16;       // local col, tile 0
    const int nl1 = nl0 + 16;            // local col, tile 1
    const int n0g = nblk * 64 + nl0;     // global col, tile 0
    const int n1g = n0g + 16;

    // Per-lane invariants (n is fixed per lane for the whole kernel)
    float wih0[K_IN], wih1[K_IN];
    #pragma unroll
    for (int i = 0; i < K_IN; ++i) {
        wih0[i] = Wih[n0g * K_IN + i];
        wih1[i] = Wih[n1g * K_IN + i];
    }
    const float bias0 = b_ih[n0g] + b_hh[n0g];
    const float bias1 = b_ih[n1g] + b_hh[n1g];
    const float fw0 = fc_w[n0g], fw1 = fc_w[n1g];

    const int mA  = m0 + l16;            // A-fragment row this lane loads
    const int sw0 = (nl0 & 15) << 3;
    const int sw1 = (nl1 & 15) << 3;
    const _Float16* sWr0 = sW + (nl0 << 9);
    const _Float16* sWr1 = sW + (nl1 << 9);

    // Ping-pong via pointer swap (PHI of two addrspace(1) args -> global_load/store,
    // NOT flat: keeps LOADcnt decoupled from DScnt and off the LDS path).
    _Float16* hin  = h0;
    _Float16* hout = h1;

    for (int t = 0; t < SEQ; ++t) {
        // ---- C init: x[t] @ W_ih^T + (b_ih+b_hh), K=6, direct into WMMA accum layout ----
        v8f c0, c1;
        const float* xt = x + (size_t)t * BATCH * K_IN;
        #pragma unroll
        for (int r = 0; r < 8; ++r) {
            int mc = m0 + hsel * 8 + r;
            const float* xr = xt + mc * K_IN;
            float a0 = bias0, a1 = bias1;
            #pragma unroll
            for (int i = 0; i < K_IN; ++i) {
                float xv = xr[i];
                a0 = fmaf(xv, wih0[i], a0);
                a1 = fmaf(xv, wih1[i], a1);
            }
            c0[r] = a0; c1[r] = a1;
        }

        // ---- h @ W_hh^T : K=512 in 16 WMMA steps of 32 ----
        const _Float16* hrow = hin + (size_t)mA * DIM;
        #pragma unroll
        for (int ks = 0; ks < 16; ++ks) {
            const int kk = ks * 32;
            // A fragment: elems 0-7 -> K=ka.., elems 8-15 -> K=ka+16..
            const int ka = kk + hsel * 8;
            v8h aLo = *(const v8h*)(hrow + ka);
            v8h aHi = *(const v8h*)(hrow + ka + 16);
            v16h A;
            #pragma unroll
            for (int i = 0; i < 8; ++i) { A[i] = aLo[i]; A[i + 8] = aHi[i]; }
            // B fragment: elems 0-7 -> K=kb.., elems 8-15 -> K=kb+16.. (swizzled 8-blocks in LDS)
            const int kb = kk + hsel * 8;
            v8h b0lo = *(const v8h*)(sWr0 + (kb ^ sw0));
            v8h b0hi = *(const v8h*)(sWr0 + ((kb + 16) ^ sw0));
            v8h b1lo = *(const v8h*)(sWr1 + (kb ^ sw1));
            v8h b1hi = *(const v8h*)(sWr1 + ((kb + 16) ^ sw1));
            v16h B0, B1;
            #pragma unroll
            for (int i = 0; i < 8; ++i) {
                B0[i] = b0lo[i]; B0[i + 8] = b0hi[i];
                B1[i] = b1lo[i]; B1[i + 8] = b1hi[i];
            }
            c0 = __builtin_amdgcn_wmma_f32_16x16x32_f16(false, A, false, B0,
                                                        (short)0, c0, false, false);
            c1 = __builtin_amdgcn_wmma_f32_16x16x32_f16(false, A, false, B1,
                                                        (short)0, c1, false, false);
        }

        // ---- tanh, store next h, accumulate y partials ----
        #pragma unroll
        for (int r = 0; r < 8; ++r) {
            int mc = m0 + hsel * 8 + r;
            float t0 = fast_tanh(c0[r]);
            float t1 = fast_tanh(c1[r]);
            hout[(size_t)mc * DIM + n0g] = (_Float16)t0;
            hout[(size_t)mc * DIM + n1g] = (_Float16)t1;
            if (t == SEQ - 1) {
                out[SEQ * BATCH + (size_t)mc * DIM + n0g] = t0;
                out[SEQ * BATCH + (size_t)mc * DIM + n1g] = t1;
            }
            float p = fmaxf(t0, 0.0f) * fw0 + fmaxf(t1, 0.0f) * fw1;
            // reduce over the 16 N-lanes of this half
            #pragma unroll
            for (int off = 1; off < 16; off <<= 1)
                p += __shfl_xor(p, off, 32);
            if (l16 == 0)
                atomicAdd(&out[t * BATCH + mc], p);
        }

        // ---- device-wide split barrier: per-step counter (no reset races) ----
        __syncthreads();
        if (tid == 0) {
            __threadfence();                 // h stores visible at device scope
            atomicAdd(&cnt[t], 1u);
            while (__hip_atomic_load(&cnt[t], __ATOMIC_ACQUIRE,
                                     __HIP_MEMORY_SCOPE_AGENT) < (unsigned)NWG) {
                __builtin_amdgcn_s_sleep(2);
            }
            __threadfence();
        }
        __syncthreads();

        // swap ping-pong buffers
        _Float16* tmp = hin; hin = hout; hout = tmp;
    }
}

extern "C" void kernel_launch(void* const* d_in, const int* in_sizes, int n_in,
                              void* d_out, int out_size, void* d_ws, size_t ws_size,
                              hipStream_t stream) {
    const float* x      = (const float*)d_in[0];
    const float* hidden = (const float*)d_in[1];
    const float* Wih    = (const float*)d_in[2];
    const float* Whh    = (const float*)d_in[3];
    const float* bih    = (const float*)d_in[4];
    const float* bhh    = (const float*)d_in[5];
    const float* fcw    = (const float*)d_in[6];
    const float* fcb    = (const float*)d_in[7];
    float* out = (float*)d_out;

    // workspace: two f16 h ping-pong buffers + per-step barrier counters (~520 KB)
    _Float16* h0  = (_Float16*)d_ws;
    _Float16* h1  = h0 + BATCH * DIM;
    unsigned* cnt = (unsigned*)(h1 + BATCH * DIM);

    rnn_init<<<(SEQ * BATCH) / TPB, TPB, 0, stream>>>(out, fcb, hidden, h0, cnt);
    rnn_persistent<<<NWG, TPB, 0, stream>>>(x, Wih, Whh, bih, bhh, fcw, out, h0, h1, cnt);
}
